// SwinEncoder_3599182594905
// MI455X (gfx1250) — compile-verified, matrix operand order checked by reading, not run on hardware
//
#include <hip/hip_runtime.h>

// ---------------------------------------------------------------------------
// CDNA5 (gfx1250) Swin encoder forward.
// All matmuls run on v_wmma_f32_16x16x32_bf16 (bf16 in, f32 accumulate).
// GEMM stages A/B tiles through LDS with coalesced b128 loads (B transposed
// at staging so fragment reads are contiguous ds_load_b128), and prefetches
// the next k-slab with global_prefetch_b8.  wave32 everywhere.
// ---------------------------------------------------------------------------

typedef __attribute__((ext_vector_type(16))) __bf16 bf16x16;
typedef __attribute__((ext_vector_type(8)))  float  f32x8;

__device__ __forceinline__ f32x8 wmma_bf16(bf16x16 a, bf16x16 b, f32x8 c) {
  return __builtin_amdgcn_wmma_f32_16x16x32_bf16(false, a, false, b, (short)0, c,
                                                 false, false);
}

// Load a 16x32 WMMA fragment from a row-major bf16 buffer (LDS or global)
// whose rows are 16B-aligned.  Per ISA 7.12.2 layout:
//   lanes 0-15 : row = r0+lane,    K = {k0..k0+7, k0+16..k0+23}
//   lanes 16-31: row = r0+lane-16, K = {k0+8..k0+15, k0+24..k0+31}
// Both 8-element chunks are contiguous -> two 16B vector loads.
__device__ __forceinline__ bf16x16 frag_ld(const __bf16* S, int ld, int r0, int k0,
                                           int lane) {
  int r  = r0 + (lane & 15);
  int kb = k0 + ((lane >> 4) << 3);
  const __bf16* p = S + (size_t)r * ld + kb;
  union { int4 q[2]; bf16x16 v; } u;
  u.q[0] = *(const int4*)p;
  u.q[1] = *(const int4*)(p + 16);
  return u.v;
}

// ---------------------------------------------------------------------------
// Tiled GEMM: C[M,N] = A[M,K]_bf16 * B[K,N]_bf16 (+ bias) (optional GELU)
// Block tile 128x32, 8 waves; wave w owns rows [16w,16w+16) x all 32 cols
// (two 16x16 accumulators).  A staged 128x32 in LDS (coalesced b128 loads),
// B staged 32x32 transposed in LDS so fragments read contiguously.
// Requires M % 128 == 0, N % 32 == 0, K % 32 == 0 (true for every call here).
// flags: bit0 = exact GELU epilogue, bit1 = write bf16 instead of f32
// ---------------------------------------------------------------------------
__global__ __launch_bounds__(256) void k_gemm(const __bf16* __restrict__ A,
                                              const __bf16* __restrict__ Bm,
                                              const float* __restrict__ bias,
                                              void* __restrict__ Cout, int M, int N,
                                              int K, int flags) {
  __shared__ alignas(16) __bf16 sA[128 * 32];   // 8 KB
  __shared__ alignas(16) __bf16 sBT[32 * 32];   // 2 KB, [n][k] transposed

  int tid = threadIdx.x, lane = tid & 31, wv = tid >> 5;
  int nblk = N >> 5;
  int m0 = (blockIdx.x / nblk) * 128;
  int n0 = (blockIdx.x % nblk) * 32;

  int arow = tid >> 1, ahalf = tid & 1;          // A staging map (256 ld)
  int brow = tid >> 2, bq = tid & 3;             // B staging map (128 ld)

  f32x8 acc0 = {}, acc1 = {};
  for (int k0 = 0; k0 < K; k0 += 32) {
    __syncthreads();
    // stage A[128x32]: one b128 per thread, coalesced along K
    *(int4*)(sA + arow * 32 + ahalf * 16) =
        *(const int4*)(A + (size_t)(m0 + arow) * K + k0 + ahalf * 16);
    // stage B[32x32] transposed: threads 0..127 read b128 along N, scatter to LDS
    if (tid < 128) {
      int4 vb = *(const int4*)(Bm + (size_t)(k0 + brow) * N + n0 + bq * 8);
      const __bf16* e = (const __bf16*)&vb;
#pragma unroll
      for (int j = 0; j < 8; ++j) sBT[(bq * 8 + j) * 32 + brow] = e[j];
    }
    // prefetch next k-slab while WMMAs run on this one
    if (k0 + 32 < K) {
      __builtin_prefetch(A + (size_t)(m0 + arow) * K + (k0 + 32) + ahalf * 16, 0, 1);
      if (tid < 128)
        __builtin_prefetch(Bm + (size_t)(k0 + 32 + brow) * N + n0 + bq * 8, 0, 1);
    }
    __syncthreads();
    bf16x16 a  = frag_ld(sA, 32, wv * 16, 0, lane);
    bf16x16 b0 = frag_ld(sBT, 32, 0, 0, lane);
    bf16x16 b1 = frag_ld(sBT, 32, 16, 0, lane);
    acc0 = wmma_bf16(a, b0, acc0);
    acc1 = wmma_bf16(a, b1, acc1);
  }

  int mb = m0 + wv * 16 + ((lane >> 4) << 3);
#pragma unroll
  for (int t = 0; t < 2; ++t) {
    const f32x8& acc = t ? acc1 : acc0;
    int n = n0 + t * 16 + (lane & 15);
    float bv = bias ? bias[n] : 0.0f;
#pragma unroll
    for (int r = 0; r < 8; ++r) {
      float v = acc[r] + bv;
      if (flags & 1) v = 0.5f * v * (1.0f + erff(v * 0.70710678118654752f));
      if (flags & 2)
        ((__bf16*)Cout)[(size_t)(mb + r) * N + n] = (__bf16)v;
      else
        ((float*)Cout)[(size_t)(mb + r) * N + n] = v;
    }
  }
}

// ---------------------------------------------------------------------------
// LayerNorm: one wave per row.
// ---------------------------------------------------------------------------
__global__ __launch_bounds__(256) void k_layernorm(const float* __restrict__ in,
                                                   const float* __restrict__ g,
                                                   const float* __restrict__ b,
                                                   float* __restrict__ out, int rows,
                                                   int C) {
  int row  = (int)((blockIdx.x * 256 + threadIdx.x) >> 5);
  int lane = threadIdx.x & 31;
  if (row >= rows) return;
  const float* x = in + (size_t)row * C;
  float s = 0.0f;
  for (int c = lane; c < C; c += 32) s += x[c];
#pragma unroll
  for (int off = 16; off; off >>= 1) s += __shfl_xor(s, off, 32);
  float mean = s / C;
  float v = 0.0f;
  for (int c = lane; c < C; c += 32) {
    float d = x[c] - mean;
    v += d * d;
  }
#pragma unroll
  for (int off = 16; off; off >>= 1) v += __shfl_xor(v, off, 32);
  float inv = rsqrtf(v / C + 1e-5f);
  float* o = out + (size_t)row * C;
  for (int c = lane; c < C; c += 32) o[c] = (x[c] - mean) * inv * g[c] + b[c];
}

// ---------------------------------------------------------------------------
// Elementwise helpers
// ---------------------------------------------------------------------------
__global__ void k_f32_to_bf16(const float* __restrict__ in, __bf16* __restrict__ out,
                              size_t n) {
  size_t i = (size_t)blockIdx.x * blockDim.x + threadIdx.x;
  if (i < n) out[i] = (__bf16)in[i];
}

__global__ void k_add(float* __restrict__ x, const float* __restrict__ y, size_t n) {
  size_t i = (size_t)blockIdx.x * blockDim.x + threadIdx.x;
  if (i < n) x[i] += y[i];
}

// patch weights (E=96, 3,4,4) -> bf16 B[64,96], K padded 48->64
__global__ void k_patchw(const float* __restrict__ w, __bf16* __restrict__ out) {
  int i = blockIdx.x * blockDim.x + threadIdx.x;
  if (i >= 64 * 96) return;
  int k = i / 96, n = i % 96;
  out[i] = (k < 48) ? (__bf16)w[n * 48 + k] : (__bf16)0.0f;
}

// im2col for 4x4/s4 patch embed: bf16 A[B*128*128, 64]
__global__ void k_im2col(const float* __restrict__ x, __bf16* __restrict__ out) {
  size_t idx = (size_t)blockIdx.x * blockDim.x + threadIdx.x;
  size_t total = (size_t)8 * 128 * 128 * 64;
  if (idx >= total) return;
  int k = (int)(idx & 63);
  size_t row = idx >> 6;
  if (k >= 48) { out[idx] = (__bf16)0.0f; return; }
  int c = k >> 4, rem = k & 15, iy = rem >> 2, ix = rem & 3;
  int b  = (int)(row >> 14);
  int p  = (int)(row & 16383);
  int ph = p >> 7, pw = p & 127;
  out[idx] = (__bf16)x[(((size_t)b * 3 + c) * 512 + ph * 4 + iy) * 512 + pw * 4 + ix];
}

// token layout f32 -> window layout bf16, with cyclic shift (roll by -shift)
__global__ void k_win_gather(const float* __restrict__ x, __bf16* __restrict__ out,
                             int B, int res, int C, int shift) {
  size_t idx = (size_t)blockIdx.x * blockDim.x + threadIdx.x;
  size_t total = (size_t)B * res * res * C;
  if (idx >= total) return;
  int c = (int)(idx % C);
  size_t t = idx / C;
  int tok = (int)(t & 63);
  int win = (int)(t >> 6);
  int nW  = res >> 3;
  int b   = win / (nW * nW);
  int wi  = win % (nW * nW);
  int wy  = wi / nW, wx = wi % nW;
  int hy  = wy * 8 + (tok >> 3), hx = wx * 8 + (tok & 7);
  int sy  = (hy + shift) % res, sx = (hx + shift) % res;
  out[idx] = (__bf16)x[(((size_t)b * res + sy) * res + sx) * C + c];
}

// window layout f32 -> token layout (reverse roll), residual add into x
__global__ void k_win_scatter_add(const float* __restrict__ wbuf,
                                  float* __restrict__ x, int B, int res, int C,
                                  int shift) {
  size_t idx = (size_t)blockIdx.x * blockDim.x + threadIdx.x;
  size_t total = (size_t)B * res * res * C;
  if (idx >= total) return;
  int c = (int)(idx % C);
  size_t t = idx / C;
  int w  = (int)(t % res);
  int h  = (int)((t / res) % res);
  int b  = (int)(t / ((size_t)res * res));
  int ry = (h - shift + res) % res, rx = (w - shift + res) % res;
  int nW = res >> 3;
  int win = b * nW * nW + (ry >> 3) * nW + (rx >> 3);
  int tok = (ry & 7) * 8 + (rx & 7);
  x[idx] += wbuf[((size_t)win * 64 + tok) * C + c];
}

// 2x2 patch merge gather: (B,res,res,C) -> (B*res2*res2, 4C) f32
__global__ void k_merge(const float* __restrict__ x, float* __restrict__ out, int B,
                        int res, int C) {
  int res2 = res >> 1;
  size_t total = (size_t)B * res2 * res2 * 4 * C;
  size_t idx = (size_t)blockIdx.x * blockDim.x + threadIdx.x;
  if (idx >= total) return;
  int C4 = 4 * C;
  int c4 = (int)(idx % C4);
  size_t t = idx / C4;
  int j = (int)(t % res2); t /= res2;
  int i = (int)(t % res2);
  int b = (int)(t / res2);
  int q = c4 / C, c = c4 % C;
  int dy = (q == 1 || q == 3) ? 1 : 0;  // order: (0,0),(1,0),(0,1),(1,1)
  int dx = (q >= 2) ? 1 : 0;
  out[idx] = x[(((size_t)b * res + 2 * i + dy) * res + 2 * j + dx) * C + c];
}

// ---------------------------------------------------------------------------
// Windowed attention: one block per (window, head), 4 waves.
// head_dim = 32 everywhere, N = 64 tokens per window.
// Q, K row-major in LDS; V staged transposed so P*V B-fragments are
// contiguous; probabilities staged bf16 for the second WMMA.
// ---------------------------------------------------------------------------
__device__ __forceinline__ int regcode(int p, int res) {
  return p < res - 8 ? 0 : (p < res - 4 ? 1 : 2);
}

__global__ __launch_bounds__(128) void k_attn(const float* __restrict__ qkv,
                                              const float* __restrict__ rpb,
                                              float* __restrict__ outbuf, int nh,
                                              int C, int res, int shift) {
  __shared__ alignas(16) __bf16 sQ[64 * 32];
  __shared__ alignas(16) __bf16 sK[64 * 32];
  __shared__ alignas(16) __bf16 sVT[32 * 64];   // [d][token]
  __shared__ alignas(16) float  sS[64 * 64];
  __shared__ alignas(16) __bf16 sP[64 * 64];

  int win = blockIdx.x / nh;
  int h   = blockIdx.x % nh;
  int tid = threadIdx.x, lane = tid & 31, wv = tid >> 5;
  size_t rowBase = (size_t)win * 64;
  int c3 = 3 * C;

  // stage Q (pre-scaled), K, V^T as bf16 in LDS
  for (int i = tid; i < 64 * 32; i += 128) {
    int n = i >> 5, d = i & 31;
    const float* r = qkv + (rowBase + n) * c3 + h * 32 + d;
    sQ[i] = (__bf16)(r[0] * 0.17677669529663687f);  // hd^-0.5, hd=32
    sK[i] = (__bf16)r[C];
    sVT[d * 64 + n] = (__bf16)r[2 * C];
  }
  __syncthreads();

  int nW = res >> 3;
  int wi = win % (nW * nW);
  int wy = wi / nW, wx = wi % nW;

  // S = Q K^T : wave wv owns row-tile wv, loops over 4 col-tiles
  for (int ct = 0; ct < 4; ++ct) {
    bf16x16 a = frag_ld(sQ, 32, 16 * wv, 0, lane);
    // B[k][n] = K[n][k] -> row-major K tile read with the A index pattern
    bf16x16 b = frag_ld(sK, 32, 16 * ct, 0, lane);
    f32x8 acc = {};
    acc = wmma_bf16(a, b, acc);
    int n  = 16 * ct + (lane & 15);
    int mb = 16 * wv + ((lane >> 4) << 3);
#pragma unroll
    for (int r = 0; r < 8; ++r) {
      int m  = mb + r;
      int dy = (m >> 3) - (n >> 3) + 7;
      int dx = (m & 7) - (n & 7) + 7;
      float v = acc[r] + rpb[(dy * 15 + dx) * nh + h];
      if (shift) {
        int lm = regcode(wy * 8 + (m >> 3), res) * 3 + regcode(wx * 8 + (m & 7), res);
        int lk = regcode(wy * 8 + (n >> 3), res) * 3 + regcode(wx * 8 + (n & 7), res);
        if (lm != lk) v -= 100.0f;
      }
      sS[m * 64 + n] = v;
    }
  }
  __syncthreads();

  // softmax per row, write bf16 probabilities
  if (tid < 64) {
    float* row = sS + tid * 64;
    float mx = -3.4e38f;
    for (int j = 0; j < 64; ++j) mx = fmaxf(mx, row[j]);
    float s = 0.0f;
    for (int j = 0; j < 64; ++j) {
      float e = __expf(row[j] - mx);
      row[j] = e;
      s += e;
    }
    float inv = 1.0f / s;
    for (int j = 0; j < 64; ++j) sP[tid * 64 + j] = (__bf16)(row[j] * inv);
  }
  __syncthreads();

  // O = P V : wave wv owns row-tile wv, 2 col-tiles (hd=32), K-loop over 64
  for (int nt = 0; nt < 2; ++nt) {
    f32x8 acc = {};
#pragma unroll
    for (int k0 = 0; k0 < 64; k0 += 32) {
      bf16x16 a = frag_ld(sP, 64, 16 * wv, k0, lane);
      bf16x16 b = frag_ld(sVT, 64, 16 * nt, k0, lane);  // V^T rows are d
      acc = wmma_bf16(a, b, acc);
    }
    int n  = 16 * nt + (lane & 15);
    int mb = 16 * wv + ((lane >> 4) << 3);
#pragma unroll
    for (int r = 0; r < 8; ++r)
      outbuf[(rowBase + mb + r) * C + h * 32 + n] = acc[r];
  }
}

// ---------------------------------------------------------------------------
// Host orchestration
// ---------------------------------------------------------------------------
extern "C" void kernel_launch(void* const* d_in, const int* in_sizes, int n_in,
                              void* d_out, int out_size, void* d_ws, size_t ws_size,
                              hipStream_t stream) {
  (void)in_sizes; (void)n_in; (void)out_size; (void)ws_size;
  const int B = 8;
  static const int RES[4] = {128, 64, 32, 16};
  static const int CH[4]  = {96, 192, 384, 768};
  static const int NH[4]  = {3, 6, 12, 24};
  // JAX pytree (sorted dict keys) flattening of params; d_in[0] = x.
  // block leaves: fc1_b,fc1_w,fc2_b,fc2_w,ln1_b,ln1_g,ln2_b,ln2_g,
  //               proj_b,proj_w,qkv_b,qkv_w,rpb  (13 per block)
  static const int BLK_BASE[4]  = {1, 30, 59, 88};
  static const int DOWN_BASE[3] = {27, 56, 85};   // nb, ng, red_w
  const int I_NORM_B = 114, I_NORM_G = 115;
  const int I_PATCH_B = 116, I_PATCH_NB = 117, I_PATCH_NG = 118, I_PATCH_W = 119;

  auto P = [&](int i) { return (const float*)d_in[i]; };
  const float* xin = (const float*)d_in[0];

  // ---- workspace carve ----
  char* wsp = (char*)d_ws;
  auto carve = [&](size_t bytes) {
    char* p = wsp;
    wsp += (bytes + 255) & ~(size_t)255;
    return p;
  };
  float*  WX  = (float*)carve(131072ull * 96 * 4);    // activations (token layout)
  float*  WO  = (float*)carve(131072ull * 96 * 4);    // f32 scratch (rows*C / rows2*4C)
  float*  WG  = (float*)carve(131072ull * 288 * 4);   // f32 scratch (rows*3C)
  __bf16* WA  = (__bf16*)carve(131072ull * 96 * 2);   // bf16 A (rows*C / rows2*4C)
  __bf16* WA2 = (__bf16*)carve(131072ull * 384 * 2);  // bf16 MLP hidden (rows*4C)
  __bf16* WW  = (__bf16*)carve(21000000ull * 2);      // bf16 weights

  auto cvt = [&](const float* src, __bf16* dst, size_t n) {
    k_f32_to_bf16<<<(unsigned)((n + 255) / 256), 256, 0, stream>>>(src, dst, n);
  };
  size_t woff = 0;
  auto cvtW = [&](const float* src, size_t n) -> const __bf16* {
    __bf16* dst = WW + woff;
    woff += n;
    cvt(src, dst, n);
    return dst;
  };

  // ---- convert all weights to bf16 ----
  const __bf16 *qkvW[4][2], *projW[4][2], *fc1W[4][2], *fc2W[4][2], *redW[3];
  for (int st = 0; st < 4; ++st) {
    size_t C = CH[st];
    for (int bi = 0; bi < 2; ++bi) {
      int pb = BLK_BASE[st] + bi * 13;
      fc1W[st][bi]  = cvtW(P(pb + 1), C * 4 * C);
      fc2W[st][bi]  = cvtW(P(pb + 3), 4 * C * C);
      projW[st][bi] = cvtW(P(pb + 9), C * C);
      qkvW[st][bi]  = cvtW(P(pb + 11), C * 3 * C);
    }
    if (st < 3) redW[st] = cvtW(P(DOWN_BASE[st] + 2), 4 * C * 2 * C);
  }
  __bf16* patchB = WW + woff;
  woff += 64 * 96;
  k_patchw<<<(64 * 96 + 255) / 256, 256, 0, stream>>>(P(I_PATCH_W), patchB);

  auto gemm = [&](const __bf16* A, const __bf16* Bm, const float* bias, void* Cv,
                  int M, int N, int K, int flags) {
    int blocks = (M / 128) * (N / 32);
    k_gemm<<<blocks, 256, 0, stream>>>(A, Bm, bias, Cv, M, N, K, flags);
  };
  auto LN = [&](const float* in, const float* g, const float* b, float* out, int rows,
               int C) {
    k_layernorm<<<(rows + 7) / 8, 256, 0, stream>>>(in, g, b, out, rows, C);
  };

  // ---- patch embed: im2col + WMMA GEMM + LN ----
  {
    size_t ic = (size_t)B * 128 * 128 * 64;
    k_im2col<<<(unsigned)((ic + 255) / 256), 256, 0, stream>>>(xin, WA);
    gemm(WA, patchB, P(I_PATCH_B), WO, B * 128 * 128, 96, 64, 0);
    LN(WO, P(I_PATCH_NG), P(I_PATCH_NB), WX, B * 128 * 128, 96);
  }

  // output tuple: (x_final, ds0, ds1, ds2, ds3)
  const size_t dsOff[4] = {1572864ull, 14155776ull, 20447232ull, 23592960ull};

  for (int st = 0; st < 4; ++st) {
    int res = RES[st], C = CH[st], nh = NH[st];
    int rows = B * res * res;
    int nW = res / 8, Bw = B * nW * nW;
    size_t nEl = (size_t)rows * C;
    unsigned gEl = (unsigned)((nEl + 255) / 256);

    // ds.append(x)
    hipMemcpyAsync((float*)d_out + dsOff[st], WX, nEl * sizeof(float),
                   hipMemcpyDeviceToDevice, stream);

    for (int bi = 0; bi < 2; ++bi) {
      int pb = BLK_BASE[st] + bi * 13;
      int shift = (bi & 1) ? 4 : 0;

      // ---- attention ----
      LN(WX, P(pb + 5), P(pb + 4), WO, rows, C);                 // ln1
      k_win_gather<<<gEl, 256, 0, stream>>>(WO, WA, B, res, C, shift);
      gemm(WA, qkvW[st][bi], P(pb + 10), WG, rows, 3 * C, C, 0); // qkv
      k_attn<<<Bw * nh, 128, 0, stream>>>(WG, P(pb + 12), WO, nh, C, res, shift);
      cvt(WO, WA, nEl);
      gemm(WA, projW[st][bi], P(pb + 8), WG, rows, C, C, 0);     // proj
      k_win_scatter_add<<<gEl, 256, 0, stream>>>(WG, WX, B, res, C, shift);

      // ---- MLP ----
      LN(WX, P(pb + 7), P(pb + 6), WO, rows, C);                 // ln2
      cvt(WO, WA, nEl);
      gemm(WA, fc1W[st][bi], P(pb + 0), WA2, rows, 4 * C, C, 3); // fc1 + GELU -> bf16
      gemm(WA2, fc2W[st][bi], P(pb + 2), WG, rows, C, 4 * C, 0); // fc2
      k_add<<<gEl, 256, 0, stream>>>(WX, WG, nEl);
    }

    // ---- patch merging ----
    if (st < 3) {
      int db = DOWN_BASE[st];
      int res2 = res / 2, rows2 = B * res2 * res2;
      size_t nM = (size_t)rows2 * 4 * C;
      k_merge<<<(unsigned)((nM + 255) / 256), 256, 0, stream>>>(WX, WO, B, res, C);
      LN(WO, P(db + 1), P(db + 0), WG, rows2, 4 * C);
      cvt(WG, WA, nM);
      gemm(WA, redW[st], nullptr, WX, rows2, 2 * C, 4 * C, 0);
    }
  }

  // ---- final norm -> d_out[0:8*256*768] ----
  LN(WX, P(I_NORM_G), P(I_NORM_B), (float*)d_out, B * 256, 768);
}
